// enKFMLP_89464168776327
// MI455X (gfx1250) — matrix operand order repaired and checked by reading, not compiled, not run on hardware
//
#include <hip/hip_runtime.h>

// ---------------------------------------------------------------------------
// EnKF + MLP fused pipeline for MI455X (gfx1250, wave32, WMMA)
//   Kernel B: process MLP (residual) + observation MLP on 262144 rows,
//             bf16 WMMA 16x16x32 with f32 accumulate, activations in LDS.
//             Each wave handles 4 consecutive 16-row tiles to amortize the
//             per-block weight staging.
//   Kernel C: per-batch sensor MLP (collapsed over ensemble), covariances,
//             Gauss-Jordan solve, ensemble update (f32 VALU, tiny).
// ---------------------------------------------------------------------------

#define BS   1024
#define NE   256
#define DXD  32
#define DZD  16
#define DRD  64
#define ROWS (BS * NE)        // 262144
#define NTILES (ROWS / 16)    // 16384
#define TPW  4                // tiles per wave
#define NBLK (NTILES / (8 * TPW))   // 512 blocks

typedef __attribute__((ext_vector_type(16))) __bf16        v16bf;
typedef __attribute__((ext_vector_type(2)))  __bf16        v2bf;
typedef __attribute__((ext_vector_type(8)))  float          v8f;
typedef __attribute__((ext_vector_type(8)))  unsigned short v8u16;

union FragU { v8u16 h[2]; v16bf bf; };

// Scalar f32 -> bf16 via compiler fptrunc (backend selects hw cvt if present).
__device__ __forceinline__ unsigned short f2bf(float x) {
  __bf16 h = (__bf16)x;
  return __builtin_bit_cast(unsigned short, h);
}

// Pack two f32 -> two bf16 in one dword; vector init lets the backend fuse
// into v_cvt_pk_bf16_f32 when available.
__device__ __forceinline__ unsigned int pk_bf16(float a, float b) {
#if __has_builtin(__builtin_amdgcn_cvt_pk_bf16_f32)
  auto v = __builtin_amdgcn_cvt_pk_bf16_f32(a, b);
  return __builtin_bit_cast(unsigned int, v);
#else
  v2bf v = { (__bf16)a, (__bf16)b };
  return __builtin_bit_cast(unsigned int, v);
#endif
}

__device__ __forceinline__ v8f wmma_bf16(v16bf a, v16bf b, v8f c) {
  // D(16x16,f32) = A(16x32,bf16) x B(32x16,bf16) + C
  return __builtin_amdgcn_wmma_f32_16x16x32_bf16(false, a, false, b,
                                                 (short)0, c, false, false);
}

// A-matrix 16x32 bf16 fragment per ISA 7.12.2:
//   lanes 0-15 (g=0): halves 0-7 = K[kc..kc+7],  halves 8-15 = K[kc+16..kc+23]
//   lanes 16-31(g=1): halves 0-7 = K[kc+8..+15], halves 8-15 = K[kc+24..+31]
// act is LDS row-major [16][W] bf16 (stored as u16), W = Kin of this layer.
__device__ __forceinline__ v16bf load_A(const unsigned short* act, int W,
                                        int kc, int lane) {
  int g = lane >> 4, m = lane & 15;
  FragU f;
  f.h[0] = *(const v8u16*)(act + m * W + kc + 8 * g);
  f.h[1] = *(const v8u16*)(act + m * W + kc + 16 + 8 * g);
  return f.bf;
}

// B-matrix 32x16 bf16 fragment: lane l holds column n=l%16,
//   lanes 0-15: K = kc+0..7, kc+16..23 ; lanes 16-31: K = kc+8..15, kc+24..31.
// wT is LDS row-major transposed weights [N][K] bf16.
__device__ __forceinline__ v16bf load_B(const unsigned short* wT, int K,
                                        int kc, int nt, int lane) {
  int g = lane >> 4, n = lane & 15;
  const unsigned short* p = wT + (nt * 16 + n) * K + kc + 8 * g;
  FragU f;
  f.h[0] = *(const v8u16*)p;
  f.h[1] = *(const v8u16*)(p + 16);
  return f.bf;
}

// D fragment: lanes 0-15 VGPR v -> (M=v, N=lane); lanes 16-31 -> (M=v+8).
template <bool RELU>
__device__ __forceinline__ void store_D(v8f d, unsigned short* act, int W,
                                        int nt, int lane, const float* bias) {
  int g = lane >> 4;
  int n = nt * 16 + (lane & 15);
  float bb = bias[n];
#pragma unroll
  for (int v = 0; v < 8; v += 2) {
    float x0 = d[v] + bb;
    float x1 = d[v + 1] + bb;
    if (RELU) { x0 = fmaxf(x0, 0.0f); x1 = fmaxf(x1, 0.0f); }
    unsigned int u = pk_bf16(x0, x1);
    act[(v + 8 * g) * W + n]     = (unsigned short)u;
    act[(v + 1 + 8 * g) * W + n] = (unsigned short)(u >> 16);
  }
}

template <int KIN, int NOUT, bool RELU>
__device__ __forceinline__ void layer(const unsigned short* actIn,
                                      unsigned short* actOut,
                                      const unsigned short* wT,
                                      const float* bias, int lane) {
  // Hoist A fragments (tile-invariant across N-tiles of this layer).
  v16bf a[KIN / 32];
#pragma unroll
  for (int kc = 0; kc < KIN / 32; ++kc) a[kc] = load_A(actIn, KIN, kc * 32, lane);
#pragma unroll
  for (int nt = 0; nt < NOUT / 16; ++nt) {
    v8f acc = {};
#pragma unroll
    for (int kc = 0; kc < KIN / 32; ++kc) {
      v16bf b = load_B(wT, KIN, kc * 32, nt, lane);
      acc = wmma_bf16(a[kc], b, acc);
    }
    store_D<RELU>(acc, actOut, NOUT, nt, lane, bias);
  }
}

// Convert f32 weights [K][N] (row-major) -> bf16 transposed [N][K] in LDS.
__device__ __forceinline__ void cvtT(const float* W, unsigned short* dst,
                                     int K, int N, int t) {
  for (int i = t; i < K * N; i += 256) {
    int k = i / N, n = i - k * N;
    dst[n * K + k] = f2bf(W[i]);
  }
}

// LDS offsets (in bf16 elements) for transposed weights
#define OPW1 0        // 32x32  -> 1024
#define OPW2 1024     // 64x32  -> 2048
#define OPW3 3072     // 64x64  -> 4096
#define OPW4 7168     // 32x64  -> 2048
#define OPW5 9216     // 32x32  -> 1024
#define OOW1 10240    // 1024
#define OOW2 11264    // 2048
#define OOW3 13312    // 4096
#define OOW4 17408    // 2048
#define OOW5 19456    // 16x32 -> 512
#define WTOT 19968
// bias offsets (f32)
#define BPB1 0
#define BPB2 32
#define BPB3 96
#define BPB4 160
#define BPB5 192
#define BOB1 224
#define BOB2 256
#define BOB3 320
#define BOB4 384
#define BOB5 416
#define BTOT 432

__global__ void __launch_bounds__(256)
enkf_mlp_wmma(const float* __restrict__ state,
              const float* pw1, const float* pb1, const float* pw2, const float* pb2,
              const float* pw3, const float* pb3, const float* pw4, const float* pb4,
              const float* pw5, const float* pb5,
              const float* ow1, const float* ob1, const float* ow2, const float* ob2,
              const float* ow3, const float* ob3, const float* ow4, const float* ob4,
              const float* ow5, const float* ob5,
              float* __restrict__ xpw, float* __restrict__ hxw) {
  __shared__ __align__(32) unsigned short sW[WTOT];
  __shared__ float sB[BTOT];
  __shared__ __align__(32) unsigned short sAct[8 * 2 * 16 * 64];  // 32 KB

  const int t = threadIdx.x;
  const int lane = t & 31;
  const int wv = t >> 5;

  // Stage transposed bf16 weights + f32 biases into LDS (shared by 8 waves).
  cvtT(pw1, sW + OPW1, 32, 32, t);
  cvtT(pw2, sW + OPW2, 32, 64, t);
  cvtT(pw3, sW + OPW3, 64, 64, t);
  cvtT(pw4, sW + OPW4, 64, 32, t);
  cvtT(pw5, sW + OPW5, 32, 32, t);
  cvtT(ow1, sW + OOW1, 32, 32, t);
  cvtT(ow2, sW + OOW2, 32, 64, t);
  cvtT(ow3, sW + OOW3, 64, 64, t);
  cvtT(ow4, sW + OOW4, 64, 32, t);
  cvtT(ow5, sW + OOW5, 32, 16, t);
  if (t < 32)  sB[BPB1 + t] = pb1[t];
  if (t < 64)  sB[BPB2 + t] = pb2[t];
  if (t < 64)  sB[BPB3 + t] = pb3[t];
  if (t < 32)  sB[BPB4 + t] = pb4[t];
  if (t < 32)  sB[BPB5 + t] = pb5[t];
  if (t < 32)  sB[BOB1 + t] = ob1[t];
  if (t < 64)  sB[BOB2 + t] = ob2[t];
  if (t < 64)  sB[BOB3 + t] = ob3[t];
  if (t < 32)  sB[BOB4 + t] = ob4[t];
  if (t < 16)  sB[BOB5 + t] = ob5[t];
  __syncthreads();

  unsigned short* A0 = sAct + wv * 2048;  // 16x64 bf16
  unsigned short* A1 = A0 + 1024;
  const int tileBase = (blockIdx.x * 8 + wv) * TPW;

  for (int it = 0; it < TPW; ++it) {
    const int rowBase = (tileBase + it) * 16;
    const float* xg = state + (size_t)rowBase * DXD;
    if (it + 1 < TPW) {
      // Prefetch next tile's state rows into cache (global_prefetch_b8).
      __builtin_prefetch(xg + 16 * DXD + lane * 16, 0, 3);
    }

    // Load x0 tile [16][32] f32 -> bf16 LDS
    for (int e = lane; e < 16 * DXD; e += 32) A0[e] = f2bf(xg[e]);

    // ---- process model ----
    layer<32, 32, true>(A0, A1, sW + OPW1, sB + BPB1, lane);
    layer<32, 64, true>(A1, A0, sW + OPW2, sB + BPB2, lane);
    layer<64, 64, true>(A0, A1, sW + OPW3, sB + BPB3, lane);
    layer<64, 32, true>(A1, A0, sW + OPW4, sB + BPB4, lane);
    // L5 (32->32, no relu) + residual: xp = x0 + upd ; emit f32 Xp + bf16 act
    {
      v16bf a5 = load_A(A0, 32, 0, lane);
#pragma unroll
      for (int nt = 0; nt < 2; ++nt) {
        v8f acc = {};
        v16bf b = load_B(sW + OPW5, 32, 0, nt, lane);
        acc = wmma_bf16(a5, b, acc);
        int g = lane >> 4;
        int n = nt * 16 + (lane & 15);
        float bb = sB[BPB5 + n];
#pragma unroll
        for (int v = 0; v < 8; v += 2) {
          int m0 = v + 8 * g, m1 = m0 + 1;
          float xp0 = xg[m0 * DXD + n] + acc[v] + bb;
          float xp1 = xg[m1 * DXD + n] + acc[v + 1] + bb;
          xpw[(size_t)rowBase * DXD + m0 * DXD + n] = xp0;
          xpw[(size_t)rowBase * DXD + m1 * DXD + n] = xp1;
          unsigned int u = pk_bf16(xp0, xp1);
          A1[m0 * DXD + n] = (unsigned short)u;
          A1[m1 * DXD + n] = (unsigned short)(u >> 16);
        }
      }
    }

    // ---- observation model ----
    layer<32, 32, true>(A1, A0, sW + OOW1, sB + BOB1, lane);
    layer<32, 64, true>(A0, A1, sW + OOW2, sB + BOB2, lane);
    layer<64, 64, true>(A1, A0, sW + OOW3, sB + BOB3, lane);
    layer<64, 32, true>(A0, A1, sW + OOW4, sB + BOB4, lane);
    // O5 (32->16, no relu) -> HX f32
    {
      v8f acc = {};
      v16bf a = load_A(A1, 32, 0, lane);
      v16bf b = load_B(sW + OOW5, 32, 0, 0, lane);
      acc = wmma_bf16(a, b, acc);
      int g = lane >> 4;
      int n = lane & 15;
      float bb = sB[BOB5 + n];
#pragma unroll
      for (int v = 0; v < 8; ++v) {
        int m = v + 8 * g;
        hxw[(size_t)rowBase * DZD + m * DZD + n] = acc[v] + bb;
      }
    }
  }
}

// ---------------------------------------------------------------------------
// Kernel C: one block per batch. Sensor MLP collapses over ensemble (obs is
// tiled, so enc/y/r are identical per member and encoding-mean == enc).
// ---------------------------------------------------------------------------
__global__ void __launch_bounds__(256)
enkf_analysis(const float* __restrict__ xpw, const float* __restrict__ hxw,
              const float* __restrict__ obs,
              const float* sw1, const float* sb1, const float* sw2, const float* sb2,
              const float* sw3, const float* sb3, const float* sw4, const float* sb4,
              const float* nw, const float* nb,
              float* __restrict__ out) {
  __shared__ float sXp[NE * DXD];   // 32 KB
  __shared__ float sHX[NE * DZD];   // 16 KB
  __shared__ float h1[64], h2[32], enc[32], yv[16], rd[16];
  __shared__ float mx[DXD], mz[DZD];
  __shared__ float M[16 * 48];      // [Pzz | Pxz^T] augmented
  __shared__ float fac[16];
  __shared__ float piv;

  const int t = threadIdx.x;
  const int b = blockIdx.x;
  const float* xb = xpw + (size_t)b * NE * DXD;
  const float* hb = hxw + (size_t)b * NE * DZD;

  for (int i = t; i < NE * DXD; i += 256) sXp[i] = xb[i];
  for (int i = t; i < NE * DZD; i += 256) sHX[i] = hb[i];

  // sensor MLP (per batch, not per member)
  const float* ob = obs + b * DRD;
  if (t < 64) {
    float s = sb1[t];
    for (int k = 0; k < 64; ++k) s += ob[k] * sw1[k * 64 + t];
    h1[t] = fmaxf(s, 0.0f);
  }
  __syncthreads();
  if (t < 32) {
    float s = sb2[t];
    for (int k = 0; k < 64; ++k) s += h1[k] * sw2[k * 32 + t];
    h2[t] = fmaxf(s, 0.0f);
  }
  __syncthreads();
  if (t < 32) {
    float s = sb3[t];
    for (int k = 0; k < 32; ++k) s += h2[k] * sw3[k * 32 + t];
    enc[t] = fmaxf(s, 0.0f);
  }
  __syncthreads();
  if (t < 16) {
    float s = sb4[t];
    for (int k = 0; k < 32; ++k) s += enc[k] * sw4[k * 16 + t];
    yv[t] = s;
  } else if (t < 32) {
    int z = t - 16;
    float d = nb[z];
    for (int k = 0; k < 32; ++k) d += enc[k] * nw[k * 16 + z];
    rd[z] = d * d + 0.1f;
  }
  __syncthreads();

  // ensemble means
  if (t < 32) {
    float s = 0.0f;
    for (int n = 0; n < NE; ++n) s += sXp[n * DXD + t];
    mx[t] = s * (1.0f / NE);
  } else if (t < 48) {
    int z = t - 32;
    float s = 0.0f;
    for (int n = 0; n < NE; ++n) s += sHX[n * DZD + z];
    mz[z] = s * (1.0f / NE);
  }
  __syncthreads();
  for (int i = t; i < NE * DXD; i += 256) sXp[i] -= mx[i & (DXD - 1)];
  for (int i = t; i < NE * DZD; i += 256) sHX[i] -= mz[i & (DZD - 1)];
  __syncthreads();

  // Pzz = HA^T HA / (NE-1) + diag(r)
  {
    int z = t >> 4, w = t & 15;
    float s = 0.0f;
    for (int n = 0; n < NE; ++n) s += sHX[n * DZD + z] * sHX[n * DZD + w];
    M[z * 48 + w] = s * (1.0f / (NE - 1)) + (z == w ? rd[z] : 0.0f);
  }
  // Pxz^T[z][x] = (A^T HA)^T / (NE-1)
  for (int e = t; e < DXD * DZD; e += 256) {
    int x = e >> 4, z = e & 15;
    float s = 0.0f;
    for (int n = 0; n < NE; ++n) s += sXp[n * DXD + x] * sHX[n * DZD + z];
    M[z * 48 + 16 + x] = s * (1.0f / (NE - 1));
  }
  __syncthreads();

  // Gauss-Jordan: [Pzz | Pxz^T] -> [I | K^T]
  for (int i = 0; i < 16; ++i) {
    if (t == 0) piv = 1.0f / M[i * 48 + i];
    if (t < 16) fac[t] = M[t * 48 + i];
    __syncthreads();
    if (t < 48) M[i * 48 + t] *= piv;
    __syncthreads();
    for (int e = t; e < 16 * 48; e += 256) {
      int r = e / 48, c = e - r * 48;
      if (r != i) M[r * 48 + c] -= fac[r] * M[i * 48 + c];
    }
    __syncthreads();
  }

  // innov[n][z] = y[z] - HX[n][z] = y[z] - mz[z] - HA[n][z]  (in place)
  for (int i = t; i < NE * DZD; i += 256) {
    int z = i & (DZD - 1);
    sHX[i] = yv[z] - mz[z] - sHX[i];
  }
  __syncthreads();

  // x_new = Xp + innov @ K^T
  float* outb = out + (size_t)b * NE * DXD;
  for (int e = t; e < NE * DXD; e += 256) {
    int n = e >> 5, x = e & (DXD - 1);
    float s = sXp[e] + mx[x];
    const float* in = &sHX[n * DZD];
    for (int z = 0; z < 16; ++z) s += in[z] * M[z * 48 + 16 + x];
    outb[e] = s;
  }
}

extern "C" void kernel_launch(void* const* d_in, const int* in_sizes, int n_in,
                              void* d_out, int out_size, void* d_ws, size_t ws_size,
                              hipStream_t stream) {
  const float* state = (const float*)d_in[0];
  const float* obs   = (const float*)d_in[1];
  const float* pw1 = (const float*)d_in[2];  const float* pb1 = (const float*)d_in[3];
  const float* pw2 = (const float*)d_in[4];  const float* pb2 = (const float*)d_in[5];
  const float* pw3 = (const float*)d_in[6];  const float* pb3 = (const float*)d_in[7];
  const float* pw4 = (const float*)d_in[8];  const float* pb4 = (const float*)d_in[9];
  const float* pw5 = (const float*)d_in[10]; const float* pb5 = (const float*)d_in[11];
  const float* ow1 = (const float*)d_in[12]; const float* ob1 = (const float*)d_in[13];
  const float* ow2 = (const float*)d_in[14]; const float* ob2 = (const float*)d_in[15];
  const float* ow3 = (const float*)d_in[16]; const float* ob3 = (const float*)d_in[17];
  const float* ow4 = (const float*)d_in[18]; const float* ob4 = (const float*)d_in[19];
  const float* ow5 = (const float*)d_in[20]; const float* ob5 = (const float*)d_in[21];
  const float* sw1 = (const float*)d_in[22]; const float* sb1 = (const float*)d_in[23];
  const float* sw2 = (const float*)d_in[24]; const float* sb2 = (const float*)d_in[25];
  const float* sw3 = (const float*)d_in[26]; const float* sb3 = (const float*)d_in[27];
  const float* sw4 = (const float*)d_in[28]; const float* sb4 = (const float*)d_in[29];
  const float* nw  = (const float*)d_in[30]; const float* nb  = (const float*)d_in[31];

  float* xpw = (float*)d_ws;                    // [ROWS][32] f32, 32 MB
  float* hxw = xpw + (size_t)ROWS * DXD;        // [ROWS][16] f32, 16 MB
  float* out = (float*)d_out;

  enkf_mlp_wmma<<<NBLK, 256, 0, stream>>>(
      state, pw1, pb1, pw2, pb2, pw3, pb3, pw4, pb4, pw5, pb5,
      ow1, ob1, ow2, ob2, ow3, ob3, ow4, ob4, ow5, ob5, xpw, hxw);

  enkf_analysis<<<BS, 256, 0, stream>>>(
      xpw, hxw, obs, sw1, sb1, sw2, sb2, sw3, sb3, sw4, sb4, nw, nb, out);
}
